// RetinaNet_6983616823946
// MI455X (gfx1250) — matrix operand order repaired and checked by reading, not compiled, not run on hardware
//
#include <hip/hip_runtime.h>
#include <hip/hip_bf16.h>

// ---------------------------------------------------------------------------
// RetinaNet head for MI455X (gfx1250): implicit-GEMM conv3x3 via WMMA f16.
// - B (weight) fragments: ping-pong register buffer, loaded one tap ahead
//   (16x global_load_b128 in a clause overlapping the 8 WMMAs of a tap).
// - A (activation) fragments: depth-2 software pipeline enforced with
//   __builtin_amdgcn_sched_group_barrier (DS-read x6, then [WMMA, DS x2]x5,
//   then WMMA x3) so the pressure-driven scheduler cannot re-serialize the
//   ds_load -> s_wait_dscnt 0 -> wmma pattern.
// ---------------------------------------------------------------------------

typedef __attribute__((ext_vector_type(16))) _Float16 v16h;
typedef __attribute__((ext_vector_type(8)))  float    v8f;

#define CIN 256
#define NKB 8          // 256 / 32 k-blocks
#define NANCH 21824    // total anchors per image (4 * 5456)
#define NPOS  5456     // total spatial positions per image
#define TOPK_N 1000

union V16U { uint4 u[2]; v16h h; };

#if defined(__has_builtin)
#if __has_builtin(__builtin_amdgcn_sched_group_barrier)
#define HAVE_SGB 1
#endif
#endif
#ifndef HAVE_SGB
#define HAVE_SGB 0
#endif

#if HAVE_SGB
// Masks: 0x008 = MFMA/WMMA, 0x100 = DS read.
// Pipeline: 6 DS reads up front (3 A fragments), then each WMMA overlaps the
// DS reads of the fragment 3 ahead; tail of 3 WMMAs drains the pipe.
#define TAP_SCHEDULE()                                                       \
  do {                                                                       \
    __builtin_amdgcn_sched_group_barrier(0x100, 6, 0);                       \
    __builtin_amdgcn_sched_group_barrier(0x008, 1, 0);                       \
    __builtin_amdgcn_sched_group_barrier(0x100, 2, 0);                       \
    __builtin_amdgcn_sched_group_barrier(0x008, 1, 0);                       \
    __builtin_amdgcn_sched_group_barrier(0x100, 2, 0);                       \
    __builtin_amdgcn_sched_group_barrier(0x008, 1, 0);                       \
    __builtin_amdgcn_sched_group_barrier(0x100, 2, 0);                       \
    __builtin_amdgcn_sched_group_barrier(0x008, 1, 0);                       \
    __builtin_amdgcn_sched_group_barrier(0x100, 2, 0);                       \
    __builtin_amdgcn_sched_group_barrier(0x008, 1, 0);                       \
    __builtin_amdgcn_sched_group_barrier(0x100, 2, 0);                       \
    __builtin_amdgcn_sched_group_barrier(0x008, 3, 0);                       \
  } while (0)
#else
#define TAP_SCHEDULE() do { } while (0)
#endif

// ---------------------------------------------------------------------------
// Pack OIHW fp32 conv weights into per-lane WMMA B-fragment order (f16):
// layout [nb][tap][kb][lane][16 halfs], lane L holds N = L&15,
// K = (L>>4)*8 + j (j<8), 16 + (L>>4)*8 + j-8 (j>=8) -> one 32B chunk per lane.
// ---------------------------------------------------------------------------
__global__ __launch_bounds__(256) void pack_weights_kernel(
    const float* __restrict__ w, _Float16* __restrict__ wp, int Cout) {
  int nB = Cout >> 4;
  int total = nB * 9 * NKB * 32 * 16;
  for (int e = blockIdx.x * blockDim.x + threadIdx.x; e < total;
       e += gridDim.x * blockDim.x) {
    int j    = e & 15;
    int lane = (e >> 4) & 31;
    int kb   = (e >> 9) & 7;
    int rem  = e >> 12;
    int tap  = rem % 9;
    int nb   = rem / 9;
    int cout = nb * 16 + (lane & 15);
    int cin  = kb * 32 + (j < 8 ? (lane >> 4) * 8 + j : 16 + (lane >> 4) * 8 + (j - 8));
    wp[e] = (_Float16)w[((size_t)cout * CIN + cin) * 9 + tap];
  }
}

// ---------------------------------------------------------------------------
// NCHW fp32 -> NHWC f16 (channels contiguous => 16B-aligned staging loads).
// ---------------------------------------------------------------------------
__global__ __launch_bounds__(256) void nchw_to_nhwc_f16_kernel(
    const float* __restrict__ in, _Float16* __restrict__ out, int H, int W) {
  int total = 8 * CIN * H * W;
  for (int e = blockIdx.x * blockDim.x + threadIdx.x; e < total;
       e += gridDim.x * blockDim.x) {
    int x = e % W;
    int y = (e / W) % H;
    int c = (e / (W * H)) % CIN;
    int b = e / (W * H * CIN);
    out[(((size_t)b * H + y) * W + x) * CIN + c] = (_Float16)in[e];
  }
}

// One tap: 16 ds_load_b128 (8 A fragments) + 8 WMMAs, interleaved by the
// sched_group_barrier pipeline above.
#define TAP_COMPUTE(ap, Bbuf)                                                \
  {                                                                          \
    V16U A[NKB];                                                             \
    _Pragma("unroll")                                                        \
    for (int kb = 0; kb < NKB; ++kb) {                                       \
      A[kb].u[0] = *reinterpret_cast<const uint4*>((ap) + kb * 32);          \
      A[kb].u[1] = *reinterpret_cast<const uint4*>((ap) + kb * 32 + 16);     \
    }                                                                        \
    _Pragma("unroll")                                                        \
    for (int kb = 0; kb < NKB; ++kb) {                                       \
      acc = __builtin_amdgcn_wmma_f32_16x16x32_f16(                          \
          false, A[kb].h, false, Bbuf[kb].h, (short)0, acc, false, false);   \
    }                                                                        \
    TAP_SCHEDULE();                                                          \
  }

// ---------------------------------------------------------------------------
// Implicit-GEMM conv3x3 (SAME, stride 1), Cin=256, via V_WMMA_F32_16X16X32_F16.
// Block = 128 threads (4 waves), one (batch, 4x4 pixel tile) per block.
// Wave w handles cout blocks w, w+4, ... ; 72 WMMAs per 16x16 output tile.
//
// A fragment layout (16-bit A 16x32, ISA 7.12.2): lane L, hi = L>>4:
//   halfs j<8  -> K = hi*8 + j        (contiguous 16B in LDS, NHWC)
//   halfs j>=8 -> K = 16 + hi*8 + j-8 (contiguous 16B at +32B)
// ---------------------------------------------------------------------------
__global__ __launch_bounds__(128, 4) void conv3x3_wmma_kernel(
    const _Float16* __restrict__ in, const _Float16* __restrict__ wpack,
    const float* __restrict__ bias, int H, int W, int Cout,
    _Float16* __restrict__ outh, float* __restrict__ outf, int relu) {
  __shared__ __align__(16) _Float16 stage[36 * CIN];  // 6x6 halo, 18.4 KB

  int tid = threadIdx.x;
  int tilesW = W >> 2;
  int tx = blockIdx.x % tilesW;
  int ty = blockIdx.x / tilesW;
  int b  = blockIdx.y;

  // Stage 6x6 x 256ch halo tile into LDS (zero padded), 16B chunks.
  for (int c = tid; c < 36 * (CIN / 8); c += 128) {
    int pos = c >> 5;  // 32 chunks of 8 halfs per position
    int ch  = c & 31;
    int py  = (pos / 6) - 1 + ty * 4;
    int px  = (pos % 6) - 1 + tx * 4;
    uint4 v = make_uint4(0u, 0u, 0u, 0u);
    if (py >= 0 && py < H && px >= 0 && px < W)
      v = *reinterpret_cast<const uint4*>(
          in + ((((size_t)b * H + py) * W + px) * CIN + ch * 8));
    *reinterpret_cast<uint4*>(&stage[pos * CIN + ch * 8]) = v;
  }
  __syncthreads();

  int wave = tid >> 5, lane = tid & 31;
  int M   = lane & 15, hi = lane >> 4;
  int pyM = M >> 2,    pxM = M & 3;
  int nBlocks = Cout >> 4;

#pragma unroll 1
  for (int nb = wave; nb < nBlocks; nb += 4) {
    float bval = bias[nb * 16 + (lane & 15)];
    v8f acc;
#pragma unroll
    for (int r = 0; r < 8; ++r) acc[r] = bval;

    // Per-lane weight base: uint4 index (tap*8+kb)*64 below this.
    const uint4* wb = reinterpret_cast<const uint4*>(wpack) +
                      ((size_t)nb * 2304 + lane) * 2;  // 2304 = 9*8*32

    V16U B0[NKB], B1[NKB];
    // Prime the pipeline with tap 0 weights.
#pragma unroll
    for (int kb = 0; kb < NKB; ++kb) {
      B0[kb].u[0] = wb[kb * 64];
      B0[kb].u[1] = wb[kb * 64 + 1];
    }

#pragma unroll 1
    for (int tap = 0; tap < 9; tap += 2) {
      // Prefetch tap+1 weights into B1 while computing tap with B0.
      if (tap + 1 < 9) {
#pragma unroll
        for (int kb = 0; kb < NKB; ++kb) {
          const uint4* wk = wb + (size_t)(tap + 1) * 512 + kb * 64;
          B1[kb].u[0] = wk[0];
          B1[kb].u[1] = wk[1];
        }
      }
      {
        int dy = tap / 3, dx = tap % 3;
        const _Float16* ap =
            &stage[((pyM + dy) * 6 + (pxM + dx)) * CIN + hi * 8];
        TAP_COMPUTE(ap, B0)
      }
      if (tap + 1 < 9) {
        // Prefetch tap+2 weights into B0 while computing tap+1 with B1.
        if (tap + 2 < 9) {
#pragma unroll
          for (int kb = 0; kb < NKB; ++kb) {
            const uint4* wk = wb + (size_t)(tap + 2) * 512 + kb * 64;
            B0[kb].u[0] = wk[0];
            B0[kb].u[1] = wk[1];
          }
        }
        int t1 = tap + 1;
        int dy = t1 / 3, dx = t1 % 3;
        const _Float16* ap =
            &stage[((pyM + dy) * 6 + (pxM + dx)) * CIN + hi * 8];
        TAP_COMPUTE(ap, B1)
      }
    }

    // C/D layout: VGPR r -> M = r + hi*8, N = lane&15
    int coutBase = nb * 16 + (lane & 15);
#pragma unroll
    for (int r = 0; r < 8; ++r) {
      float v = acc[r];
      if (relu) v = fmaxf(v, 0.0f);
      int Mr = r + hi * 8;
      int oy = ty * 4 + (Mr >> 2);
      int ox = tx * 4 + (Mr & 3);
      size_t o = (((size_t)b * H + oy) * W + ox) * Cout + coutBase;
      if (outf) outf[o] = v;
      else      outh[o] = (_Float16)v;
    }
  }
}

// ---------------------------------------------------------------------------
// Decode: per (batch, anchor) -> score = max over 80 classes (>=0 else -inf),
// argmax class, box decode against anchors.
// ---------------------------------------------------------------------------
__global__ __launch_bounds__(256) void decode_kernel(
    const float* __restrict__ cls, const float* __restrict__ reg,
    const float* __restrict__ anchors, float* __restrict__ score,
    float* __restrict__ box4, int* __restrict__ clsi) {
  int g = blockIdx.x * blockDim.x + threadIdx.x;
  if (g >= 8 * NANCH) return;
  int b = g / NANCH, ga = g % NANCH;

  int HWl, aoff, poff;
  if      (ga < 16384) { HWl = 4096; aoff = 0;     poff = 0;    }
  else if (ga < 20480) { HWl = 1024; aoff = 16384; poff = 4096; }
  else if (ga < 21504) { HWl = 256;  aoff = 20480; poff = 5120; }
  else if (ga < 21760) { HWl = 64;   aoff = 21504; poff = 5376; }
  else                 { HWl = 16;   aoff = 21760; poff = 5440; }
  int wi = ga - aoff;
  int a = wi / HWl, pos = wi % HWl;

  const float* c = cls + (size_t)2560 * poff + ((size_t)b * HWl + pos) * 320 + a * 80;
  float mv = c[0];
  int   mi = 0;
  for (int k = 1; k < 80; ++k) {
    float v = c[k];
    if (v > mv) { mv = v; mi = k; }
  }
  const float* r = reg + (size_t)128 * poff + ((size_t)b * HWl + pos) * 16 + a * 4;

  float ax = anchors[ga];
  float ay = anchors[NANCH + ga];
  float aw = anchors[2 * NANCH + ga];
  float ah = anchors[3 * NANCH + ga];
  float whx = aw * r[2], why = ah * r[3];
  float xx = ax + r[0] * ax, yy = ay + r[1] * ay;
  float x1 = xx - 0.5f * whx, y1 = yy - 0.5f * why;
  float x2 = x1 + whx,        y2 = y1 + why;

  int o = b * NANCH + ga;
  score[o] = (mv >= 0.0f) ? mv : -__builtin_inff();
  box4[o * 4 + 0] = x1;
  box4[o * 4 + 1] = y1;
  box4[o * 4 + 2] = x2;
  box4[o * 4 + 3] = y2;
  clsi[o] = mi;
}

// ---------------------------------------------------------------------------
// Per-batch top-1000 via full bitonic sort of 32768 (score,idx) pairs held
// entirely in LDS (256 KB of the WGP's 320 KB). One block per batch.
// Writes boxes/scores/cls directly into d_out and a valid mask into ws.
// ---------------------------------------------------------------------------
__global__ __launch_bounds__(1024) void topk_sort_kernel(
    const float* __restrict__ score, const float* __restrict__ box4,
    const int* __restrict__ clsi, float* __restrict__ out,
    int* __restrict__ validw) {
  extern __shared__ unsigned char smem[];
  float* sk = reinterpret_cast<float*>(smem);
  int*   sv = reinterpret_cast<int*>(smem + 32768 * sizeof(float));
  const int N = 32768;
  int b = blockIdx.x, tid = threadIdx.x;
  float ninf = -__builtin_inff();

  for (int i = tid; i < N; i += blockDim.x) {
    if (i < NANCH) { sk[i] = score[(size_t)b * NANCH + i]; sv[i] = i; }
    else           { sk[i] = ninf;                         sv[i] = -1; }
  }
  __syncthreads();

  for (int size = 2; size <= N; size <<= 1) {
    for (int stride = size >> 1; stride > 0; stride >>= 1) {
      for (int i = tid; i < N; i += blockDim.x) {
        int j = i ^ stride;
        if (j > i) {
          float a = sk[i], c = sk[j];
          bool lower = ((i & size) == 0);
          bool sw = lower ? (a < c) : (a > c);  // overall descending
          if (sw) {
            sk[i] = c; sk[j] = a;
            int t = sv[i]; sv[i] = sv[j]; sv[j] = t;
          }
        }
      }
      __syncthreads();
    }
  }

  for (int i = tid; i < TOPK_N; i += blockDim.x) {
    float s = sk[i];
    int  id = sv[i];
    bool valid = (s > ninf);
    out[32000 + b * TOPK_N + i] = valid ? s : 0.0f;
    float b0 = 0.f, b1 = 0.f, b2 = 0.f, b3 = 0.f;
    int cv = 0;
    if (id >= 0) {
      const float* bp = &box4[((size_t)b * NANCH + id) * 4];
      b0 = bp[0]; b1 = bp[1]; b2 = bp[2]; b3 = bp[3];
      cv = clsi[(size_t)b * NANCH + id];
    }
    float* ob = out + ((size_t)b * TOPK_N + i) * 4;
    ob[0] = b0; ob[1] = b1; ob[2] = b2; ob[3] = b3;
    reinterpret_cast<int*>(out)[40000 + b * TOPK_N + i] = cv;
    validw[b * TOPK_N + i] = valid ? 1 : 0;
  }
}

// ---------------------------------------------------------------------------
// NMS, matching the reference's sequential suppression semantics exactly:
// keep = valid; for i: if keep[i]: suppress j>i with IoU > 0.5.
// One block per batch; boxes + keep flags live in LDS.
// ---------------------------------------------------------------------------
__global__ __launch_bounds__(1024) void nms_kernel(
    float* __restrict__ out, const int* __restrict__ validw) {
  __shared__ float bx[TOPK_N * 4];
  __shared__ int   kp[TOPK_N];
  int b = blockIdx.x, tid = threadIdx.x;

  for (int i = tid; i < TOPK_N; i += blockDim.x) {
    const float* p = out + ((size_t)b * TOPK_N + i) * 4;
    bx[i * 4 + 0] = p[0]; bx[i * 4 + 1] = p[1];
    bx[i * 4 + 2] = p[2]; bx[i * 4 + 3] = p[3];
    kp[i] = validw[b * TOPK_N + i];
  }
  __syncthreads();

  for (int i = 0; i < TOPK_N; ++i) {
    if (kp[i]) {  // uniform: all threads read same LDS word
      float x1 = bx[i * 4], y1 = bx[i * 4 + 1];
      float x2 = bx[i * 4 + 2], y2 = bx[i * 4 + 3];
      float ar = fmaxf(x2 - x1, 0.f) * fmaxf(y2 - y1, 0.f);
      for (int j = i + 1 + tid; j < TOPK_N; j += blockDim.x) {
        if (kp[j]) {
          float u1 = fmaxf(x1, bx[j * 4]),     v1 = fmaxf(y1, bx[j * 4 + 1]);
          float u2 = fminf(x2, bx[j * 4 + 2]), v2 = fminf(y2, bx[j * 4 + 3]);
          float inter = fmaxf(u2 - u1, 0.f) * fmaxf(v2 - v1, 0.f);
          float arj = fmaxf(bx[j * 4 + 2] - bx[j * 4], 0.f) *
                      fmaxf(bx[j * 4 + 3] - bx[j * 4 + 1], 0.f);
          float iou = inter / (ar + arj - inter + 1e-9f);
          if (iou > 0.5f) kp[j] = 0;
        }
      }
    }
    __syncthreads();
  }

  for (int i = tid; i < TOPK_N; i += blockDim.x)
    out[48000 + b * TOPK_N + i] = kp[i] ? 1.0f : 0.0f;
}

// ---------------------------------------------------------------------------
// Host orchestration.
// ---------------------------------------------------------------------------
extern "C" void kernel_launch(void* const* d_in, const int* in_sizes, int n_in,
                              void* d_out, int out_size, void* d_ws,
                              size_t ws_size, hipStream_t stream) {
  (void)in_sizes; (void)n_in; (void)out_size; (void)ws_size;
  const float* p[5] = {(const float*)d_in[0], (const float*)d_in[1],
                       (const float*)d_in[2], (const float*)d_in[3],
                       (const float*)d_in[4]};
  const float* cls_w  = (const float*)d_in[5];
  const float* cls_b  = (const float*)d_in[6];
  const float* cls_ow = (const float*)d_in[7];
  const float* cls_ob = (const float*)d_in[8];
  const float* reg_w  = (const float*)d_in[9];
  const float* reg_b  = (const float*)d_in[10];
  const float* reg_ow = (const float*)d_in[11];
  const float* reg_ob = (const float*)d_in[12];
  const float* anchors = (const float*)d_in[13];

  unsigned char* ws = (unsigned char*)d_ws;
  size_t off = 0;
  auto alloc = [&](size_t bytes) -> void* {
    void* pp = (void*)(ws + off);
    off += (bytes + 255) & ~(size_t)255;
    return pp;
  };

  const size_t innerW = (size_t)16 * 9 * NKB * 32 * 16;  // 589824 halfs
  _Float16* wp_cls[4];
  _Float16* wp_reg[4];
  for (int i = 0; i < 4; ++i) wp_cls[i] = (_Float16*)alloc(innerW * 2);
  for (int i = 0; i < 4; ++i) wp_reg[i] = (_Float16*)alloc(innerW * 2);
  _Float16* wp_clso = (_Float16*)alloc((size_t)20 * 9 * NKB * 512 * 2);
  _Float16* wp_rego = (_Float16*)alloc((size_t)1 * 9 * NKB * 512 * 2);

  const size_t actBytes = (size_t)8 * 4096 * CIN * 2;  // largest level, f16
  _Float16* xin = (_Float16*)alloc(actBytes);
  _Float16* b0  = (_Float16*)alloc(actBytes);
  _Float16* b1  = (_Float16*)alloc(actBytes);

  float* clsout = (float*)alloc((size_t)8 * NPOS * 320 * 4);
  float* regout = (float*)alloc((size_t)8 * NPOS * 16 * 4);
  float* score  = (float*)alloc((size_t)8 * NANCH * 4);
  float* box4   = (float*)alloc((size_t)8 * NANCH * 4 * 4);
  int*   clsi   = (int*)alloc((size_t)8 * NANCH * 4);
  int*   validw = (int*)alloc((size_t)8 * TOPK_N * 4);

  // Pack all conv weights into WMMA B-fragment order (f16).
  {
    int tot256 = 16 * 9 * NKB * 512;
    int g256 = (tot256 + 255) / 256;
    for (int i = 0; i < 4; ++i) {
      pack_weights_kernel<<<g256, 256, 0, stream>>>(
          cls_w + (size_t)i * 589824, wp_cls[i], 256);
      pack_weights_kernel<<<g256, 256, 0, stream>>>(
          reg_w + (size_t)i * 589824, wp_reg[i], 256);
    }
    int tot320 = 20 * 9 * NKB * 512;
    pack_weights_kernel<<<(tot320 + 255) / 256, 256, 0, stream>>>(
        cls_ow, wp_clso, 320);
    int tot16 = 1 * 9 * NKB * 512;
    pack_weights_kernel<<<(tot16 + 255) / 256, 256, 0, stream>>>(
        reg_ow, wp_rego, 16);
  }

  const int Hs[5] = {64, 32, 16, 8, 4};
  const int poff[5] = {0, 4096, 5120, 5376, 5440};

  for (int l = 0; l < 5; ++l) {
    int H = Hs[l], W = Hs[l];
    int HW = H * W;
    int totIn = 8 * CIN * HW;
    nchw_to_nhwc_f16_kernel<<<(totIn + 255) / 256, 256, 0, stream>>>(
        p[l], xin, H, W);

    dim3 grid((unsigned)((H / 4) * (W / 4)), 8);
    float* clsO = clsout + (size_t)2560 * poff[l];
    float* regO = regout + (size_t)128 * poff[l];

    // cls tower: 4x (conv+relu) -> output conv (320 ch, f32)
    conv3x3_wmma_kernel<<<grid, 128, 0, stream>>>(xin, wp_cls[0], cls_b + 0,   H, W, 256, b0, nullptr, 1);
    conv3x3_wmma_kernel<<<grid, 128, 0, stream>>>(b0,  wp_cls[1], cls_b + 256, H, W, 256, b1, nullptr, 1);
    conv3x3_wmma_kernel<<<grid, 128, 0, stream>>>(b1,  wp_cls[2], cls_b + 512, H, W, 256, b0, nullptr, 1);
    conv3x3_wmma_kernel<<<grid, 128, 0, stream>>>(b0,  wp_cls[3], cls_b + 768, H, W, 256, b1, nullptr, 1);
    conv3x3_wmma_kernel<<<grid, 128, 0, stream>>>(b1,  wp_clso,   cls_ob,      H, W, 320, nullptr, clsO, 0);

    // reg tower: 4x (conv+relu) -> output conv (16 ch, f32)
    conv3x3_wmma_kernel<<<grid, 128, 0, stream>>>(xin, wp_reg[0], reg_b + 0,   H, W, 256, b0, nullptr, 1);
    conv3x3_wmma_kernel<<<grid, 128, 0, stream>>>(b0,  wp_reg[1], reg_b + 256, H, W, 256, b1, nullptr, 1);
    conv3x3_wmma_kernel<<<grid, 128, 0, stream>>>(b1,  wp_reg[2], reg_b + 512, H, W, 256, b0, nullptr, 1);
    conv3x3_wmma_kernel<<<grid, 128, 0, stream>>>(b0,  wp_reg[3], reg_b + 768, H, W, 256, b1, nullptr, 1);
    conv3x3_wmma_kernel<<<grid, 128, 0, stream>>>(b1,  wp_rego,   reg_ob,      H, W, 16,  nullptr, regO, 0);
  }

  int totA = 8 * NANCH;
  decode_kernel<<<(totA + 255) / 256, 256, 0, stream>>>(
      clsout, regout, anchors, score, box4, clsi);

  topk_sort_kernel<<<8, 1024, 32768 * 8, stream>>>(
      score, box4, clsi, (float*)d_out, validw);

  nms_kernel<<<8, 1024, 0, stream>>>((float*)d_out, validw);
}
